// KNRM_TF_Class_76922864271367
// MI455X (gfx1250) — matrix operand order repaired and checked by reading, not compiled, not run on hardware
//
#include <hip/hip_runtime.h>
#include <hip/hip_bf16.h>
#include <math.h>

typedef __attribute__((ext_vector_type(16))) __bf16 v16bf;
typedef __attribute__((ext_vector_type(8)))  __bf16 v8bf;
typedef __attribute__((ext_vector_type(4)))  __bf16 v4bf;
typedef __attribute__((ext_vector_type(8)))  float  v8f;

#define QLEN   16
#define DLEN   1024
#define EMB    300     // = 75 float4 per row, rows 16B aligned (1200 B)
#define KPAD   320     // K padded to multiple of 32 for 16x16x32 WMMA (80 x v4bf)
#define NK     11
#define LSTR   328     // LDS row stride (elements); 328*2B -> conflict-free 16B frag loads
#define NCOLS  128     // doc columns per block (4 waves x 2 tiles x 16)
#define NWAVES 4

__device__ __constant__ float c_mu[NK] =
    {-0.9f,-0.7f,-0.5f,-0.3f,-0.1f,0.1f,0.3f,0.5f,0.7f,0.9f,1.0f};
// 0.5 / sigma^2
__device__ __constant__ float c_hs[NK] =
    {50.f,50.f,50.f,50.f,50.f,50.f,50.f,50.f,50.f,50.f,500000.f};

__global__ void knrm_zero(float* __restrict__ p, int n) {
    int i = blockIdx.x * 256 + threadIdx.x;
    if (i < n) p[i] = 0.0f;
}

__global__ __launch_bounds__(128)
void knrm_gemm_pool(const int* __restrict__ doc, const int* __restrict__ query,
                    const float* __restrict__ emb,
                    float* __restrict__ kern_g, float* __restrict__ rowsum_g) {
    __shared__ __bf16 sQ[QLEN  * LSTR];   // normalized query tile (bf16)
    __shared__ __bf16 sD[NCOLS * LSTR];   // normalized doc tile   (bf16)

    const int b     = blockIdx.y;
    const int nbase = blockIdx.x * NCOLS;       // doc-column tile base
    const int wid   = threadIdx.x >> 5;
    const int lane  = threadIdx.x & 31;

    // ---- Phase 1: gather + L2-normalize rows into LDS as bf16; PAD(0) rows -> 0 ----
    // One wave per row: 3x global_load_b128 (75 float4), register-resident, then
    // 3x ds_store_b64 of packed bf16 with zero padding for K=300..319.
    for (int row = wid; row < QLEN + NCOLS; row += NWAVES) {
        int tok; __bf16* dst;
        if (row < QLEN) { tok = query[b * QLEN + row];              dst = &sQ[row * LSTR]; }
        else            { tok = doc[b * DLEN + nbase + row - QLEN]; dst = &sD[(row - QLEN) * LSTR]; }
        const float4* e4 = (const float4*)(emb + (long long)tok * EMB);
        const int i2 = lane + 64;
        float4 x0 = e4[lane];                                  // i0 = 0..31  < 75
        float4 x1 = e4[lane + 32];                             // i1 = 32..63 < 75
        float4 x2 = (i2 < 75) ? e4[i2] : make_float4(0.f, 0.f, 0.f, 0.f);
        float ss = x0.x*x0.x + x0.y*x0.y + x0.z*x0.z + x0.w*x0.w
                 + x1.x*x1.x + x1.y*x1.y + x1.z*x1.z + x1.w*x1.w
                 + x2.x*x2.x + x2.y*x2.y + x2.z*x2.z + x2.w*x2.w;
        #pragma unroll
        for (int off = 16; off; off >>= 1) ss += __shfl_xor(ss, off, 32);
        const float scale = (tok != 0) ? 1.0f / (sqrtf(ss) + 1e-9f) : 0.0f;
        v4bf p0 = { (__bf16)(x0.x*scale), (__bf16)(x0.y*scale),
                    (__bf16)(x0.z*scale), (__bf16)(x0.w*scale) };
        v4bf p1 = { (__bf16)(x1.x*scale), (__bf16)(x1.y*scale),
                    (__bf16)(x1.z*scale), (__bf16)(x1.w*scale) };
        *(v4bf*)&dst[4 * lane]        = p0;
        *(v4bf*)&dst[4 * (lane + 32)] = p1;
        if (i2 < 80) {  // covers K up to 320; x2==0 for i2>=75 -> zero padding
            v4bf p2 = { (__bf16)(x2.x*scale), (__bf16)(x2.y*scale),
                        (__bf16)(x2.z*scale), (__bf16)(x2.w*scale) };
            *(v4bf*)&dst[4 * i2] = p2;
        }
    }
    __syncthreads();

    // ---- Phase 2: two 16x16 sim tiles per wave, v_wmma_f32_16x16x32_bf16, K=320 ----
    const int nt0  = wid * 32;        // this wave's first doc-row tile inside sD
    const int mrow = lane & 15;       // A: row M / B: column N for this lane
    const int half = lane >> 4;
    v8f c0 = {}, c1 = {};
    #pragma unroll
    for (int kc = 0; kc < KPAD; kc += 32) {
        // A fragment (16-bit A 16x32 layout): lanes 0-15 K=[kc..+7]+[kc+16..+23],
        // lanes 16-31 K=[kc+8..+15]+[kc+24..+31]; shared by both WMMAs.
        const int ka = kc + half * 8;
        v8bf a0 = *(const v8bf*)&sQ[mrow * LSTR + ka];
        v8bf a1 = *(const v8bf*)&sQ[mrow * LSTR + ka + 16];
        // B fragments (16-bit B 32x16 layout): lanes 0-15 K=[kc..+7]+[kc+16..+23], lanes 16-31 K=[kc+8..+15]+[kc+24..+31]
        const int kb = kc + half * 8;
        const __bf16* r0 = &sD[(nt0 + mrow) * LSTR + kb];
        const __bf16* r1 = &sD[(nt0 + 16 + mrow) * LSTR + kb];
        v8bf b00 = *(const v8bf*)r0;
        v8bf b01 = *(const v8bf*)(r0 + 16);
        v8bf b10 = *(const v8bf*)r1;
        v8bf b11 = *(const v8bf*)(r1 + 16);
        v16bf A, B0, B1;
        #pragma unroll
        for (int i = 0; i < 8; ++i) {
            A[i]  = a0[i];  A[i + 8]  = a1[i];
            B0[i] = b00[i]; B0[i + 8] = b01[i];
            B1[i] = b10[i]; B1[i + 8] = b11[i];
        }
        c0 = __builtin_amdgcn_wmma_f32_16x16x32_bf16(false, A, false, B0,
                                                     (short)0, c0, false, false);
        c1 = __builtin_amdgcn_wmma_f32_16x16x32_bf16(false, A, false, B1,
                                                     (short)0, c1, false, false);
    }

    // ---- Phase 3: RBF kernel bank over both tiles, butterfly over doc columns ----
    // D layout: lane holds sim[q = r + 8*half][col = base + (lane&15)] in cX[r]
    #pragma unroll
    for (int r = 0; r < 8; ++r) {
        const int q  = r + 8 * half;
        const float s0 = c0[r];
        const float s1 = c1[r];
        float v[NK + 1];
        v[0] = s0 + s1;                             // rowsum contribution (mask source)
        #pragma unroll
        for (int kk = 0; kk < NK; ++kk) {
            const float d0 = s0 - c_mu[kk];
            const float d1 = s1 - c_mu[kk];
            v[kk + 1] = __expf(-d0 * d0 * c_hs[kk]) + __expf(-d1 * d1 * c_hs[kk]);
        }
        // butterfly over the 16 doc columns; strides 1..8 stay inside each half-wave
        #pragma unroll
        for (int off = 1; off <= 8; off <<= 1) {
            #pragma unroll
            for (int j = 0; j < NK + 1; ++j) v[j] += __shfl_xor(v[j], off, 32);
        }
        if ((lane & 15) == 0) {
            atomicAdd(&rowsum_g[b * QLEN + q], v[0]);
            #pragma unroll
            for (int kk = 0; kk < NK; ++kk)
                atomicAdd(&kern_g[(b * NK + kk) * QLEN + q], v[kk + 1]);
        }
    }
}

__global__ void knrm_final(const float* __restrict__ kern_g,
                           const float* __restrict__ rowsum_g,
                           const float* __restrict__ Wc,
                           const float* __restrict__ bc,
                           float* __restrict__ out) {
    const int b = blockIdx.x;
    const int lane = threadIdx.x;  // 32 threads
    float acc = 0.0f;
    for (int i = lane; i < NK * QLEN; i += 32) {
        const int kk = i / QLEN, q = i % QLEN;
        const float rs = rowsum_g[b * QLEN + q];
        const float val = (rs != 0.0f)
            ? __logf(kern_g[(b * NK + kk) * QLEN + q] + 1e-6f) : 0.0f;
        acc += val * Wc[kk];
    }
    #pragma unroll
    for (int off = 16; off; off >>= 1) acc += __shfl_xor(acc, off, 32);
    if (lane == 0) out[b] = acc + bc[0];
}

extern "C" void kernel_launch(void* const* d_in, const int* in_sizes, int n_in,
                              void* d_out, int out_size, void* d_ws, size_t ws_size,
                              hipStream_t stream) {
    const int*   doc   = (const int*)d_in[0];    // [B, 1024]
    const int*   query = (const int*)d_in[1];    // [B, 16]
    /* d_in[2] = query_idf (unused by reference) */
    const float* emb   = (const float*)d_in[3];  // [100000, 300]
    const float* Wc    = (const float*)d_in[4];  // [11, 1]
    const float* bc    = (const float*)d_in[5];  // [1]

    const int B = in_sizes[0] / DLEN;

    float* kern   = (float*)d_ws;                 // [B, 11, 16]
    float* rowsum = kern + (size_t)B * NK * QLEN; // [B, 16]
    const int nz = B * NK * QLEN + B * QLEN;

    knrm_zero<<<(nz + 255) / 256, 256, 0, stream>>>((float*)d_ws, nz);
    knrm_gemm_pool<<<dim3(DLEN / NCOLS, B), 128, 0, stream>>>(doc, query, emb, kern, rowsum);
    knrm_final<<<B, 32, 0, stream>>>(kern, rowsum, Wc, bc, (float*)d_out);
}
